// AdditiveAttention_59373627900520
// MI455X (gfx1250) — matrix operand order repaired and checked by reading, not compiled, not run on hardware
//
#include <hip/hip_runtime.h>
#include <hip/hip_bf16.h>

// ---------------------------------------------------------------------------
// AdditiveAttention on MI455X (gfx1250):
//   bf16 WMMA GEMMs, async global->LDS double-buffered staging, streaming
//   softmax pooling. Weights pre-transposed to [N][K] so A and B tiles stage
//   identically (raw 16B chunks) and all fragments load as ds_load_b128.
// ---------------------------------------------------------------------------

typedef __attribute__((ext_vector_type(16))) __bf16 v16bf;
typedef __attribute__((ext_vector_type(8)))  float  v8f;
typedef __attribute__((ext_vector_type(4)))  int    v4i;

#define DIM    1024
#define HEADS  16
#define DHEAD  64
#define NSEQ   8192
#define BATCH  4
#define LDQKV  3072
#define ATTN_SCALE 0.125f   // 64^-0.5

// Async global->LDS support (gfx1250). Guarded so host pass / toolchains
// without the builtin fall back to synchronous staging.
#if defined(__has_builtin)
#  if __has_builtin(__builtin_amdgcn_global_load_async_to_lds_b128)
#    define HAVE_ASYNC_LDS 1
#  endif
#endif
#ifndef HAVE_ASYNC_LDS
#  define HAVE_ASYNC_LDS 0
#endif

#if HAVE_ASYNC_LDS
typedef __attribute__((address_space(1))) v4i GV4;   // global int4
typedef __attribute__((address_space(3))) v4i LV4;   // LDS int4
#endif

union ABFrag { v16bf v; unsigned int u[8]; };
union V4     { uint4 u4; unsigned int u[4]; unsigned short s[8]; };

__device__ inline unsigned short f2bf(float f) {
  unsigned int u = __float_as_uint(f);
  u += 0x7fffu + ((u >> 16) & 1u);          // round-to-nearest-even
  return (unsigned short)(u >> 16);
}
__device__ inline float bf2f(unsigned short h) {
  return __uint_as_float(((unsigned int)h) << 16);
}

__device__ inline void storeOut(float* p, float v)          { *p = v; }
__device__ inline void storeOut(unsigned short* p, float v) { *p = f2bf(v); }

// copy 16B global -> LDS (async on gfx1250, sync fallback elsewhere)
__device__ inline void lds_copy16(const unsigned short* g, unsigned short* l) {
#if HAVE_ASYNC_LDS
  __builtin_amdgcn_global_load_async_to_lds_b128(
      (GV4*)g, (LV4*)l, 0, 0);
#else
  *(uint4*)l = *(const uint4*)g;
#endif
}

__device__ inline void wait_async_lds() {
#if HAVE_ASYNC_LDS
#  if __has_builtin(__builtin_amdgcn_s_wait_asynccnt)
  __builtin_amdgcn_s_wait_asynccnt(0);
#  else
  asm volatile("s_wait_asynccnt 0x0" ::: "memory");
#  endif
#endif
}

// ---------------------------------------------------------------------------
// f32 -> bf16 conversion (4 elems / thread)
// ---------------------------------------------------------------------------
__global__ __launch_bounds__(256) void cvt_f32_to_bf16(
    const float* __restrict__ in, unsigned short* __restrict__ out, long long n) {
  long long i = ((long long)blockIdx.x * blockDim.x + threadIdx.x) * 4;
  if (i + 3 < n) {
    float4 f = *(const float4*)(in + i);
    unsigned int lo = (unsigned int)f2bf(f.x) | ((unsigned int)f2bf(f.y) << 16);
    unsigned int hi = (unsigned int)f2bf(f.z) | ((unsigned int)f2bf(f.w) << 16);
    *(uint2*)(out + i) = make_uint2(lo, hi);
  }
}

// ---------------------------------------------------------------------------
// f32 [K][N] -> bf16 transposed [N][K] (32x32 LDS tile, 32x8 threads)
// ---------------------------------------------------------------------------
__global__ __launch_bounds__(256) void cvt_transpose_bf16(
    const float* __restrict__ in, unsigned short* __restrict__ out, int K, int N) {
  __shared__ unsigned short t[32][33];
  const int n0 = blockIdx.x * 32, k0 = blockIdx.y * 32;
  const int tx = threadIdx.x & 31, ty = threadIdx.x >> 5;
#pragma unroll
  for (int i = 0; i < 4; ++i) {
    int kk = ty + i * 8;
    t[kk][tx] = f2bf(in[(size_t)(k0 + kk) * N + n0 + tx]);
  }
  __syncthreads();
#pragma unroll
  for (int i = 0; i < 4; ++i) {
    int nn = ty + i * 8;
    out[(size_t)(n0 + nn) * K + k0 + tx] = t[tx][nn];
  }
}

// ---------------------------------------------------------------------------
// Tiled bf16 GEMM: C[M,N] = A[M,K] * BT[N,K]^T (+bias), f32 accumulate.
// Block tile 128x128, BK=32, 256 threads = 8 waves (4 along M x 2 along N),
// each wave 32x64 = 2x4 fragments. Double-buffered async LDS staging.
// LDS tile pitch = 40 ushorts (80B): 16B-aligned rows, bank-spread.
// ---------------------------------------------------------------------------
#define TP 40   // tile pitch in ushorts

template <typename OutT>
__global__ __launch_bounds__(256) void gemm_bf16(
    const unsigned short* __restrict__ A, const unsigned short* __restrict__ BT,
    OutT* __restrict__ C, const float* __restrict__ bias,
    int M, int N, int K, int lda, int ldb, int ldc) {
  __shared__ unsigned short As[2][128 * TP];
  __shared__ unsigned short Bs[2][128 * TP];

  const int tid    = threadIdx.x;
  const int lane   = tid & 31;
  const int wave   = tid >> 5;
  const int blockM = blockIdx.y * 128;
  const int blockN = blockIdx.x * 128;
  const int wm     = (wave >> 1) * 32;
  const int wn     = (wave & 1) * 64;
  const int mrow   = lane & 15;
  const int hihalf = lane >> 4;
  const int aKoff  = hihalf * 16;               // A layout: upper lanes K+=16 (contiguous)
  const int bKoff  = hihalf * 16;               // B layout: upper lanes K+=16 (contiguous, same as A)

  // each thread stages 2 chunks of 16B per tile (128 rows x 4 chunks = 512)
  auto stage = [&](const unsigned short* __restrict__ src, int lds, unsigned short* dst,
                   int base, int k0) {
#pragma unroll
    for (int c = 0; c < 2; ++c) {
      int ch = tid + c * 256;
      int r = ch >> 2, q = (ch & 3) * 8;
      lds_copy16(&src[(size_t)(base + r) * lds + k0 + q], &dst[r * TP + q]);
    }
  };

  v8f acc[2][4] = {};

  stage(A, lda, As[0], blockM, 0);
  stage(BT, ldb, Bs[0], blockN, 0);
  wait_async_lds();
  __syncthreads();

  int cur = 0;
  for (int k0 = 0; k0 < K; k0 += 32) {
    const int nxt = cur ^ 1;
    if (k0 + 32 < K) {                       // prefetch next tiles (async)
      stage(A, lda, As[nxt], blockM, k0 + 32);
      stage(BT, ldb, Bs[nxt], blockN, k0 + 32);
    }

    // ---- A fragments: 2x ds_load_b128 each --------------------------------
    ABFrag a[2];
#pragma unroll
    for (int fm = 0; fm < 2; ++fm) {
      const unsigned short* ap = &As[cur][(wm + fm * 16 + mrow) * TP + aKoff];
      a[fm].u[0] = *(const unsigned int*)(ap + 0);
      a[fm].u[1] = *(const unsigned int*)(ap + 2);
      a[fm].u[2] = *(const unsigned int*)(ap + 4);
      a[fm].u[3] = *(const unsigned int*)(ap + 6);
      a[fm].u[4] = *(const unsigned int*)(ap + 8);
      a[fm].u[5] = *(const unsigned int*)(ap + 10);
      a[fm].u[6] = *(const unsigned int*)(ap + 12);
      a[fm].u[7] = *(const unsigned int*)(ap + 14);
    }
    // ---- B fragments (contiguous K pairs): 2x ds_load_b128 each -----------
#pragma unroll
    for (int fn = 0; fn < 4; ++fn) {
      ABFrag bf;
      const unsigned short* bp = &Bs[cur][(wn + fn * 16 + mrow) * TP + bKoff];
#pragma unroll
      for (int v = 0; v < 8; ++v) bf.u[v] = *(const unsigned int*)(bp + 2 * v);
#pragma unroll
      for (int fm = 0; fm < 2; ++fm)
        acc[fm][fn] = __builtin_amdgcn_wmma_f32_16x16x32_bf16(
            false, a[fm].v, false, bf.v, (short)0, acc[fm][fn], false, false);
    }

    wait_async_lds();       // next tiles landed while we computed
    __syncthreads();
    cur = nxt;
  }

  // ---- epilogue -----------------------------------------------------------
#pragma unroll
  for (int fm = 0; fm < 2; ++fm)
#pragma unroll
    for (int fn = 0; fn < 4; ++fn) {
      int n = blockN + wn + fn * 16 + mrow;
      float bv = bias ? bias[n] : 0.0f;
#pragma unroll
      for (int r = 0; r < 8; ++r) {
        int m = blockM + wm + fm * 16 + r + hihalf * 8;  // C/D VGPR layout
        storeOut(&C[(size_t)m * ldc + n], acc[fm][fn][r] + bv);
      }
    }
}

// ---------------------------------------------------------------------------
// Softmax pooling: out[bh, d] = sum_n softmax_n(row_n . wmod) * row_n[d]
// One workgroup per (b,h); logits live in LDS (32 KB of 320 KB).
// ---------------------------------------------------------------------------
template <bool USE_MOD>
__global__ __launch_bounds__(256) void pool_softmax(
    const unsigned short* __restrict__ qkv, int secoff,
    const float* __restrict__ logitw, const float* __restrict__ modvec,
    float* __restrict__ outvec) {
  __shared__ float sL[NSEQ];
  __shared__ float sW[DHEAD];
  __shared__ float sR[256];

  const int tid = threadIdx.x;
  const int bh  = blockIdx.x;
  const int b   = bh >> 4, h = bh & 15;
  const size_t rowbase = (size_t)b * NSEQ;
  const int colbase = secoff + h * DHEAD;

  if (tid < DHEAD) {
    float m = USE_MOD ? modvec[bh * DHEAD + tid] : 1.0f;
    sW[tid] = logitw[tid] * m * ATTN_SCALE;
  }
  __syncthreads();

  float lmax = -3.4e38f;
  for (int n = tid; n < NSEQ; n += 256) {
    const unsigned short* row = qkv + (rowbase + n) * LDQKV + colbase;
    float dot = 0.0f;
#pragma unroll
    for (int j = 0; j < 8; ++j) {
      V4 v; v.u4 = *(const uint4*)(row + j * 8);
#pragma unroll
      for (int e = 0; e < 8; ++e) dot += bf2f(v.s[e]) * sW[j * 8 + e];
    }
    sL[n] = dot;
    lmax = fmaxf(lmax, dot);
  }
  sR[tid] = lmax; __syncthreads();
  for (int off = 128; off > 0; off >>= 1) {
    if (tid < off) sR[tid] = fmaxf(sR[tid], sR[tid + off]);
    __syncthreads();
  }
  const float gmax = sR[0];
  __syncthreads();

  float lsum = 0.0f;
  for (int n = tid; n < NSEQ; n += 256) {
    float e = __expf(sL[n] - gmax);
    sL[n] = e;
    lsum += e;
  }
  sR[tid] = lsum; __syncthreads();
  for (int off = 128; off > 0; off >>= 1) {
    if (tid < off) sR[tid] += sR[tid + off];
    __syncthreads();
  }
  const float inv = 1.0f / sR[0];
  __syncthreads();

  const int d = tid & 63;
  const int chunk = tid >> 6;
  float acc = 0.0f;
  for (int n = chunk * (NSEQ / 4); n < (chunk + 1) * (NSEQ / 4); ++n)
    acc += sL[n] * bf2f(qkv[(rowbase + n) * LDQKV + colbase + d]);
  sR[tid] = acc; __syncthreads();
  if (tid < 64)
    outvec[bh * DHEAD + tid] =
        (sR[tid] + sR[tid + 64] + sR[tid + 128] + sR[tid + 192]) * inv;
}

// ---------------------------------------------------------------------------
// Per-head: r = (v * global_k) @ w_r + b_r + q, written bf16 in [b,n,h*d].
// Grid (64 bh, 64 row-groups); 8 waves, each wave one 16x64 tile (8 WMMAs).
// ---------------------------------------------------------------------------
__global__ __launch_bounds__(256) void head_r_kernel(
    const unsigned short* __restrict__ qkv,
    const float* __restrict__ w_r, const float* __restrict__ b_r,
    const float* __restrict__ gk, unsigned short* __restrict__ r_out) {
  __shared__ unsigned short wrT[64 * 72];       // w_r^T bf16, pitch 72 (16B align)
  __shared__ unsigned short uL[8 * 16 * 64];    // per-wave u staging
  __shared__ float sBr[64];
  __shared__ float sGk[64];

  const int tid = threadIdx.x;
  const int bh  = blockIdx.x;
  const int b   = bh >> 4, h = bh & 15;

  for (int i = tid; i < 64 * 64; i += 256) {
    int kk = i >> 6, nn = i & 63;
    wrT[nn * 72 + kk] = f2bf(w_r[kk * 64 + nn]);
  }
  if (tid < 64) { sBr[tid] = b_r[tid]; sGk[tid] = gk[bh * 64 + tid]; }
  __syncthreads();

  const int wave = tid >> 5, lane = tid & 31;
  const size_t row0 = (size_t)b * NSEQ + (size_t)(blockIdx.y * 8 + wave) * 16;
  const int vcol = 2 * DIM + h * 64;
  unsigned short* u = &uL[wave * 1024];

#pragma unroll
  for (int i = 0; i < 32; ++i) {
    int idx = i * 32 + lane;
    int rr = idx >> 6, cc = idx & 63;
    float uf = bf2f(qkv[(row0 + rr) * LDQKV + vcol + cc]) * sGk[cc];
    u[rr * 64 + cc] = f2bf(uf);
  }

  const int mrow   = lane & 15;
  const int hihalf = lane >> 4;
  v8f acc[4] = {};
#pragma unroll
  for (int s = 0; s < 2; ++s) {                 // K = 64 = 2 x 32
    ABFrag a;
    const unsigned short* ap = &u[mrow * 64 + s * 32 + hihalf * 16];
    a.u[0] = *(const unsigned int*)(ap + 0);
    a.u[1] = *(const unsigned int*)(ap + 2);
    a.u[2] = *(const unsigned int*)(ap + 4);
    a.u[3] = *(const unsigned int*)(ap + 6);
    a.u[4] = *(const unsigned int*)(ap + 8);
    a.u[5] = *(const unsigned int*)(ap + 10);
    a.u[6] = *(const unsigned int*)(ap + 12);
    a.u[7] = *(const unsigned int*)(ap + 14);
#pragma unroll
    for (int fn = 0; fn < 4; ++fn) {
      ABFrag bf;
      const unsigned short* bp = &wrT[(fn * 16 + mrow) * 72 + s * 32 + hihalf * 16];
#pragma unroll
      for (int v = 0; v < 8; ++v) bf.u[v] = *(const unsigned int*)(bp + 2 * v);
      acc[fn] = __builtin_amdgcn_wmma_f32_16x16x32_bf16(
          false, a.v, false, bf.v, (short)0, acc[fn], false, false);
    }
  }

#pragma unroll
  for (int fn = 0; fn < 4; ++fn) {
    int cc = fn * 16 + mrow;
#pragma unroll
    for (int r = 0; r < 8; ++r) {
      size_t row = row0 + r + hihalf * 8;
      float val = acc[fn][r] + sBr[cc] + bf2f(qkv[row * LDQKV + h * 64 + cc]);
      r_out[row * DIM + h * 64 + cc] = f2bf(val);
    }
  }
}

// ---------------------------------------------------------------------------
// Host-side launch
// ---------------------------------------------------------------------------
extern "C" void kernel_launch(void* const* d_in, const int* in_sizes, int n_in,
                              void* d_out, int out_size, void* d_ws, size_t ws_size,
                              hipStream_t stream) {
  (void)in_sizes; (void)n_in; (void)out_size; (void)ws_size;
  const float* x    = (const float*)d_in[0];
  const float* wqkv = (const float*)d_in[1];
  const float* wql  = (const float*)d_in[2];
  const float* wkl  = (const float*)d_in[3];
  const float* wr   = (const float*)d_in[4];
  const float* br   = (const float*)d_in[5];
  const float* wout = (const float*)d_in[6];
  const float* bout = (const float*)d_in[7];
  float* out = (float*)d_out;

  char* ws = (char*)d_ws;
  unsigned short* x_bf     = (unsigned short*)(ws);                 //  64 MB
  unsigned short* wqkvT_bf = (unsigned short*)(ws + 67108864ull);   //   6 MB [3072][1024]
  unsigned short* woutT_bf = (unsigned short*)(ws + 73400320ull);   //   2 MB [1024][1024]
  unsigned short* qkv_bf   = (unsigned short*)(ws + 75497472ull);   // 192 MB
  unsigned short* r_bf     = (unsigned short*)(ws + 276824064ull);  //  64 MB
  float* g_q = (float*)(ws + 343932928ull);                         //  16 KB
  float* g_k = (float*)(ws + 343949312ull);                         //  16 KB

  const long long M = (long long)BATCH * NSEQ;                      // 32768

  // 1) demote x; transpose+demote weights to [N][K] bf16
  cvt_f32_to_bf16<<<32768, 256, 0, stream>>>(x, x_bf, M * DIM);
  cvt_transpose_bf16<<<dim3(LDQKV / 32, DIM / 32), 256, 0, stream>>>(
      wqkv, wqkvT_bf, DIM, LDQKV);
  cvt_transpose_bf16<<<dim3(DIM / 32, DIM / 32), 256, 0, stream>>>(
      wout, woutT_bf, DIM, DIM);

  // 2) qkv = x @ w_qkv   (bf16 out)
  gemm_bf16<unsigned short><<<dim3(LDQKV / 128, M / 128), 256, 0, stream>>>(
      x_bf, wqkvT_bf, qkv_bf, nullptr, (int)M, LDQKV, DIM, DIM, DIM, LDQKV);

  // 3) global_q then global_k (k-logits modulated by global_q)
  pool_softmax<false><<<BATCH * HEADS, 256, 0, stream>>>(qkv_bf, 0,   wql, nullptr, g_q);
  pool_softmax<true ><<<BATCH * HEADS, 256, 0, stream>>>(qkv_bf, DIM, wkl, g_q,     g_k);

  // 4) r = (v * global_k) @ w_r + b_r + q   (bf16 out, [b,n,h*d])
  head_r_kernel<<<dim3(BATCH * HEADS, NSEQ / 128), 256, 0, stream>>>(
      qkv_bf, wr, br, g_k, r_bf);

  // 5) out = r @ w_out + b_out  (f32 out)
  gemm_bf16<float><<<dim3(DIM / 128, M / 128), 256, 0, stream>>>(
      r_bf, woutT_bf, out, bout, (int)M, DIM, DIM, DIM, DIM, DIM);
}
